// TokenAddEraseWrite_82025285419010
// MI455X (gfx1250) — matrix operand order repaired and checked by reading, not compiled, not run on hardware
//
#include <hip/hip_runtime.h>

#define DIM 64
#define NTOK 27
#define NMEM 512
#define HID 192

typedef __attribute__((ext_vector_type(16))) _Float16 v16h;
typedef __attribute__((ext_vector_type(8)))  float    v8f;

union V16HU { v16h v; unsigned u[8]; };

// ---------------- LDS layout (bytes) ----------------
// AU:  union of LN(memory) f16 rows [512][stride 66 halfs = 132B]  and  sel f32 [512][27] (per-row 132B)
#define OFF_AU     0
#define AU_ROW     132
#define A_STRIDE_H 66
// SCR: union of per-wave hbuf f16 [8][16][194] and S1 f32[64*192] + S2 f32[64*27]
#define OFF_SCR    67584
#define HB_STRIDE  194
#define HB_WAVE_B  (16*HB_STRIDE*2)     // 6208 B per wave
#define OFF_S2     (OFF_SCR + 49152)
// WA: union of W1T f16 [192][66] and {cbuf, et, at} f32 [27*64] each
#define OFF_WA     124928
#define W1T_STRIDE 66
#define OFF_CB     OFF_WA
#define OFF_ET     (OFF_WA + 6912)
#define OFF_AT     (OFF_WA + 13824)
// WB: union of W2T f16 [32][194] and softmax reduction scratch
#define OFF_WB     150272
#define W2T_STRIDE 194
#define OFF_PMAX   OFF_WB
#define OFF_RMAX   (OFF_WB + 1728)
#define OFF_RSUM   (OFF_WB + 1840)
// biases
#define OFF_B1     162688   // f32[192]
#define OFF_B2     163456   // f32[32]
#define SMEM_TOTAL 163584

__device__ __forceinline__ float gelu_f(float x) {
    return 0.5f * x * (1.0f + erff(x * 0.70710678118654752f));
}

// A fragment (16x32 f16) from row-major f16 LDS (stride in halfs, K contiguous)
__device__ __forceinline__ v16h load_a_frag(const _Float16* base, int stride_h,
                                            int r0, int k0, int lane) {
    int m    = lane & 15;
    int koff = (lane < 16) ? 0 : 8;
    const _Float16* row = base + (r0 + m) * stride_h + k0 + koff;
    V16HU f;
#pragma unroll
    for (int i = 0; i < 4; ++i) f.u[i]     = *(const unsigned*)(row + 2 * i);
#pragma unroll
    for (int i = 0; i < 4; ++i) f.u[4 + i] = *(const unsigned*)(row + 16 + 2 * i);
    return f.v;
}

// B fragment (32x16 f16) from N-major/K-contiguous (transposed) f16 LDS
__device__ __forceinline__ v16h load_b_frag(const _Float16* base, int stride_h,
                                            int n0, int k0, int lane) {
    int n    = lane & 15;
    int koff = (lane < 16) ? 0 : 8;
    const _Float16* col = base + (n0 + n) * stride_h + k0 + koff;
    V16HU f;
#pragma unroll
    for (int i = 0; i < 4; ++i) f.u[i]     = *(const unsigned*)(col + 2 * i);
#pragma unroll
    for (int i = 0; i < 4; ++i) f.u[4 + i] = *(const unsigned*)(col + 16 + 2 * i);
    return f.v;
}

__device__ __forceinline__ v8f wmma_f16f32(v16h a, v16h b, v8f c) {
    return __builtin_amdgcn_wmma_f32_16x16x32_f16(false, a, false, b, (short)0, c,
                                                  false, false);
}

__device__ __forceinline__ float& selref(char* smem, int r, int t) {
    return *(float*)(smem + (size_t)r * AU_ROW + (size_t)t * 4);
}

// 4-stage token-mix + channel-mix MLP (scalar f32; shapes too odd for WMMA, tiny FLOPs)
__device__ void token_channel_mlp(char* smem, const float* cb,
                                  const float* W1, const float* B1,
                                  const float* W2, const float* B2,
                                  const float* P1, const float* PB1,
                                  const float* P2, const float* PB2,
                                  float* outbuf, int tid, int nth) {
    float* S1 = (float*)(smem + OFF_SCR);   // [64][192]
    float* S2 = (float*)(smem + OFF_S2);    // [64][27]
    for (int idx = tid; idx < DIM * HID; idx += nth) {
        int dim = idx / HID, h = idx % HID;
        float s = B1[h];
        for (int t = 0; t < NTOK; ++t) s += cb[t * DIM + dim] * W1[t * HID + h];
        S1[idx] = gelu_f(s);
    }
    __syncthreads();
    for (int idx = tid; idx < DIM * NTOK; idx += nth) {
        int dim = idx / NTOK, t = idx % NTOK;
        float s = B2[t];
        for (int h = 0; h < HID; ++h) s += S1[dim * HID + h] * W2[h * NTOK + t];
        S2[idx] = gelu_f(s);
    }
    __syncthreads();
    float* S3 = S1;                         // [27][192], reuses S1 region
    for (int idx = tid; idx < NTOK * HID; idx += nth) {
        int t = idx / HID, h = idx % HID;
        float s = PB1[h];
        for (int dim = 0; dim < DIM; ++dim) s += S2[dim * NTOK + t] * P1[dim * HID + h];
        S3[idx] = gelu_f(s);
    }
    __syncthreads();
    for (int idx = tid; idx < NTOK * DIM; idx += nth) {
        int t = idx / DIM, d = idx % DIM;
        float s = PB2[d];
        for (int h = 0; h < HID; ++h) s += S3[t * HID + h] * P2[h * DIM + d];
        outbuf[idx] = gelu_f(s);
    }
    __syncthreads();
}

__global__ __launch_bounds__(256, 1) void taew_kernel(
    const float* __restrict__ memory, const float* __restrict__ control,
    const float* ln1_g, const float* ln1_b,
    const float* tb1_W1, const float* tb1_b1, const float* tb1_W2, const float* tb1_b2,
    const float* lnc_g, const float* lnc_b,
    const float* tb2_W1, const float* tb2_b1, const float* tb2_W2, const float* tb2_b2,
    const float* tb2p_W1, const float* tb2p_b1, const float* tb2p_W2, const float* tb2p_b2,
    const float* tb3_W1, const float* tb3_b1, const float* tb3_W2, const float* tb3_b2,
    const float* tb3p_W1, const float* tb3p_b1, const float* tb3p_W2, const float* tb3p_b2,
    float* __restrict__ out) {
    __shared__ __align__(16) char smem[SMEM_TOTAL];
    const int tid  = threadIdx.x;
    const int lane = tid & 31;
    const int wave = tid >> 5;
    const int b    = blockIdx.x;

    _Float16* W1T  = (_Float16*)(smem + OFF_WA);
    _Float16* W2T  = (_Float16*)(smem + OFF_WB);
    float*    b1bf = (float*)(smem + OFF_B1);
    float*    b2bf = (float*)(smem + OFF_B2);

    // ---- Phase 1: stage transposed f16 weights + biases ----
    for (int idx = tid; idx < HID * DIM; idx += 256) {  // W1T[n][k] = W1[k][n]
        int n = idx >> 6, k = idx & 63;
        W1T[n * W1T_STRIDE + k] = (_Float16)tb1_W1[k * HID + n];
    }
    for (int idx = tid; idx < 32 * HID; idx += 256) {   // W2T[n][k] = W2[k][n], pad n>=27
        int n = idx / HID, k = idx % HID;
        W2T[n * W2T_STRIDE + k] = (n < NTOK) ? (_Float16)tb1_W2[k * NTOK + n] : (_Float16)0.0f;
    }
    for (int idx = tid; idx < HID; idx += 256) b1bf[idx] = tb1_b1[idx];
    if (tid < 32) b2bf[tid] = (tid < NTOK) ? tb1_b2[tid] : 0.0f;

    // ---- Phase 2: LayerNorm(memory) -> f16 A buffer ----
    for (int r = tid; r < NMEM; r += 256) {
        const float* mr = memory + ((size_t)b * NMEM + r) * DIM;
        float x[DIM];
        float s = 0.0f;
#pragma unroll
        for (int i = 0; i < 16; ++i) {
            float4 v = ((const float4*)mr)[i];
            x[4 * i] = v.x; x[4 * i + 1] = v.y; x[4 * i + 2] = v.z; x[4 * i + 3] = v.w;
            s += v.x + v.y + v.z + v.w;
        }
        float mean = s * (1.0f / DIM);
        float vs = 0.0f;
#pragma unroll
        for (int i = 0; i < DIM; ++i) { float dd = x[i] - mean; vs += dd * dd; }
        float inv = rsqrtf(vs * (1.0f / DIM) + 1e-5f);
        _Float16* dst = (_Float16*)(smem + (size_t)r * AU_ROW);
#pragma unroll
        for (int i = 0; i < DIM; ++i)
            dst[i] = (_Float16)((x[i] - mean) * inv * ln1_g[i] + ln1_b[i]);
    }
    __syncthreads();

    // ---- Phase 3: WMMA GEMM1 (16x192, K=64) + GELU -> GEMM2 (16x32, K=192) + GELU -> sel logits ----
    {
        const _Float16* Abase = (const _Float16*)smem;
        _Float16* hb = (_Float16*)(smem + OFF_SCR + wave * HB_WAVE_B);
        const int n  = lane & 15;
        const int mb = (lane < 16) ? 0 : 8;
        for (int rt = wave; rt < NMEM / 16; rt += 8) {
            const int r0 = rt * 16;
            v16h a0 = load_a_frag(Abase, A_STRIDE_H, r0, 0, lane);
            v16h a1 = load_a_frag(Abase, A_STRIDE_H, r0, 32, lane);
#pragma unroll
            for (int ct = 0; ct < HID / 16; ++ct) {
                const int n0 = ct * 16;
                v16h b0 = load_b_frag(W1T, W1T_STRIDE, n0, 0, lane);
                v16h b1 = load_b_frag(W1T, W1T_STRIDE, n0, 32, lane);
                v8f c;
#pragma unroll
                for (int i = 0; i < 8; ++i) c[i] = 0.0f;
                c = wmma_f16f32(a0, b0, c);
                c = wmma_f16f32(a1, b1, c);
                float bias = b1bf[n0 + n];
#pragma unroll
                for (int r = 0; r < 8; ++r) {
                    float v = gelu_f(c[r] + bias);
                    hb[(mb + r) * HB_STRIDE + n0 + n] = (_Float16)v;
                }
            }
            v8f c0, c1;
#pragma unroll
            for (int i = 0; i < 8; ++i) { c0[i] = 0.0f; c1[i] = 0.0f; }
#pragma unroll
            for (int kt = 0; kt < HID / 32; ++kt) {
                v16h ah  = load_a_frag(hb, HB_STRIDE, 0, kt * 32, lane);
                v16h bb0 = load_b_frag(W2T, W2T_STRIDE, 0, kt * 32, lane);
                v16h bb1 = load_b_frag(W2T, W2T_STRIDE, 16, kt * 32, lane);
                c0 = wmma_f16f32(ah, bb0, c0);
                c1 = wmma_f16f32(ah, bb1, c1);
            }
            // write sel logits (gelu applied); overwrites this wave's already-consumed LN rows
#pragma unroll
            for (int r = 0; r < 8; ++r) {
                int m = mb + r;
                selref(smem, r0 + m, n) = gelu_f(c0[r] + b2bf[n]);
                int tc = 16 + n;
                if (tc < NTOK) selref(smem, r0 + m, tc) = gelu_f(c1[r] + b2bf[tc]);
            }
        }
    }
    __syncthreads();

    // ---- Phase 4: softmax over NM per token ----
    {
        float* pmax = (float*)(smem + OFF_PMAX);
        float* rmax = (float*)(smem + OFF_RMAX);
        float* rsum = (float*)(smem + OFF_RSUM);
        for (int item = tid; item < NTOK * 16; item += 256) {
            int t = item >> 4, cc = item & 15, rb = cc * 32;
            float m = -3.4e38f;
            for (int i = 0; i < 32; ++i) m = fmaxf(m, selref(smem, rb + i, t));
            pmax[item] = m;
        }
        __syncthreads();
        if (tid < NTOK) {
            float m = pmax[tid * 16];
            for (int i = 1; i < 16; ++i) m = fmaxf(m, pmax[tid * 16 + i]);
            rmax[tid] = m;
        }
        __syncthreads();
        for (int item = tid; item < NTOK * 16; item += 256) {
            int t = item >> 4, cc = item & 15, rb = cc * 32;
            float mm = rmax[t], s = 0.0f;
            for (int i = 0; i < 32; ++i) s += __expf(selref(smem, rb + i, t) - mm);
            pmax[item] = s;
        }
        __syncthreads();
        if (tid < NTOK) {
            float s = 0.0f;
            for (int i = 0; i < 16; ++i) s += pmax[tid * 16 + i];
            rsum[tid] = s;
        }
        __syncthreads();
        for (int idx = tid; idx < NMEM * NTOK; idx += 256) {
            int r = idx / NTOK, t = idx - r * NTOK;
            selref(smem, r, t) = __expf(selref(smem, r, t) - rmax[t]) / rsum[t];
        }
    }
    __syncthreads();

    // ---- Phase 5: LayerNorm(control) -> cbuf (overlays dead W1T region) ----
    float* cb = (float*)(smem + OFF_CB);
    if (tid < NTOK) {
        const float* cr = control + ((size_t)b * NTOK + tid) * DIM;
        float s = 0.0f;
        for (int i = 0; i < DIM; ++i) s += cr[i];
        float mean = s * (1.0f / DIM);
        float vs = 0.0f;
        for (int i = 0; i < DIM; ++i) { float dd = cr[i] - mean; vs += dd * dd; }
        float inv = rsqrtf(vs * (1.0f / DIM) + 1e-5f);
        for (int i = 0; i < DIM; ++i)
            cb[tid * DIM + i] = (cr[i] - mean) * inv * lnc_g[i] + lnc_b[i];
    }
    __syncthreads();

    // ---- Phases 6/7: et and at ----
    float* etb = (float*)(smem + OFF_ET);
    float* atb = (float*)(smem + OFF_AT);
    token_channel_mlp(smem, cb, tb2_W1, tb2_b1, tb2_W2, tb2_b2,
                      tb2p_W1, tb2p_b1, tb2p_W2, tb2p_b2, etb, tid, 256);
    token_channel_mlp(smem, cb, tb3_W1, tb3_b1, tb3_W2, tb3_b2,
                      tb3p_W1, tb3p_b1, tb3p_W2, tb3p_b2, atb, tid, 256);

    // ---- Phase 8: out = mem * prod_t(1 - s*et) + (1 - sum_t s*at / NT) ----
    for (int idx = tid; idx < NMEM * DIM; idx += 256) {
        int n = idx >> 6, d = idx & 63;
        float m = memory[((size_t)b * NMEM + n) * DIM + d];
        float prod = 1.0f, acc = 0.0f;
#pragma unroll
        for (int t = 0; t < NTOK; ++t) {
            float s = selref(smem, n, t);
            prod *= (1.0f - s * etb[t * DIM + d]);
            acc  += s * atb[t * DIM + d];
        }
        out[((size_t)b * NMEM + n) * DIM + d] = m * prod + (1.0f - acc * (1.0f / NTOK));
    }
}

extern "C" void kernel_launch(void* const* d_in, const int* in_sizes, int n_in,
                              void* d_out, int out_size, void* d_ws, size_t ws_size,
                              hipStream_t stream) {
    (void)in_sizes; (void)n_in; (void)out_size; (void)d_ws; (void)ws_size;
    const float* memory  = (const float*)d_in[0];
    const float* control = (const float*)d_in[1];
    const float* ln1_g   = (const float*)d_in[2];
    const float* ln1_b   = (const float*)d_in[3];
    const float* tb1_W1  = (const float*)d_in[4];
    const float* tb1_b1  = (const float*)d_in[5];
    const float* tb1_W2  = (const float*)d_in[6];
    const float* tb1_b2  = (const float*)d_in[7];
    const float* lnc_g   = (const float*)d_in[8];
    const float* lnc_b   = (const float*)d_in[9];
    const float* tb2_W1  = (const float*)d_in[10];
    const float* tb2_b1  = (const float*)d_in[11];
    const float* tb2_W2  = (const float*)d_in[12];
    const float* tb2_b2  = (const float*)d_in[13];
    const float* tb2p_W1 = (const float*)d_in[14];
    const float* tb2p_b1 = (const float*)d_in[15];
    const float* tb2p_W2 = (const float*)d_in[16];
    const float* tb2p_b2 = (const float*)d_in[17];
    const float* tb3_W1  = (const float*)d_in[18];
    const float* tb3_b1  = (const float*)d_in[19];
    const float* tb3_W2  = (const float*)d_in[20];
    const float* tb3_b2  = (const float*)d_in[21];
    const float* tb3p_W1 = (const float*)d_in[22];
    const float* tb3p_b1 = (const float*)d_in[23];
    const float* tb3p_W2 = (const float*)d_in[24];
    const float* tb3p_b2 = (const float*)d_in[25];

    taew_kernel<<<64, 256, 0, stream>>>(
        memory, control, ln1_g, ln1_b,
        tb1_W1, tb1_b1, tb1_W2, tb1_b2, lnc_g, lnc_b,
        tb2_W1, tb2_b1, tb2_W2, tb2_b2, tb2p_W1, tb2p_b1, tb2p_W2, tb2p_b2,
        tb3_W1, tb3_b1, tb3_W2, tb3_b2, tb3p_W1, tb3p_b1, tb3p_W2, tb3p_b2,
        (float*)d_out);
}
